// MultiHeadedAttention1_4501125726582
// MI455X (gfx1250) — matrix operand order repaired and checked by reading, not compiled, not run on hardware
//
#include <hip/hip_runtime.h>

// ---------------------------------------------------------------------------
// Types for CDNA5 WMMA (wave32): v_wmma_f32_16x16x32_bf16
// ---------------------------------------------------------------------------
typedef __bf16 bf8v  __attribute__((ext_vector_type(8)));
typedef __bf16 v16bf __attribute__((ext_vector_type(16)));
typedef float  v8f   __attribute__((ext_vector_type(8)));

static __device__ __forceinline__ unsigned short f2bf(float f) {
    union { float f; unsigned u; } v; v.f = f;
    unsigned r = v.u + 0x7FFFu + ((v.u >> 16) & 1u);   // round-to-nearest-even
    return (unsigned short)(r >> 16);
}

static __device__ __forceinline__ v16bf cat16(bf8v lo, bf8v hi) {
    v16bf r;
#pragma unroll
    for (int i = 0; i < 8; ++i) { r[i] = lo[i]; r[i + 8] = hi[i]; }
    return r;
}

// A-matrix fragment (16x32 bf16): lanes 0-15 -> row=lane, K {0..7,16..23};
// lanes 16-31 -> row=lane-16, K {8..15,24..31}.  (ISA 7.12.2)
static __device__ __forceinline__ v16bf load_a_frag(const unsigned short* base,
                                                    int ld, int kbase, int lane) {
    const int row = lane & 15;
    const int k0  = kbase + ((lane & 16) ? 8 : 0);
    const unsigned short* p = base + row * ld + k0;
    return cat16(*(const bf8v*)p, *(const bf8v*)(p + 16));
}

// B-matrix fragment (32x16 bf16) from an [N][K] (K-contiguous) buffer:
// lanes 0-15 -> col=lane, K {0..7,16..23}; lanes 16-31 -> col=lane-16, K {8..15,24..31}.
static __device__ __forceinline__ v16bf load_b_frag(const unsigned short* base,
                                                    int ld, int kbase, int lane) {
    const int n  = lane & 15;
    const int k0 = kbase + ((lane & 16) ? 8 : 0);
    const unsigned short* p = base + n * ld + k0;
    return cat16(*(const bf8v*)p, *(const bf8v*)(p + 16));
}

static __device__ __forceinline__ v8f wmma_bf16(v16bf a, v16bf b, v8f c) {
    return __builtin_amdgcn_wmma_f32_16x16x32_bf16(
        false, a, false, b, (short)0, c, false, false);
}

// CDNA5 async global->LDS copy (16B per lane), tracked by ASYNCcnt.
static __device__ __forceinline__ void async_b128(unsigned lds_addr,
                                                  const unsigned short* g) {
    const unsigned long long ga = (unsigned long long)(uintptr_t)g;
    asm volatile("global_load_async_to_lds_b128 %0, %1, off"
                 :: "v"(lds_addr), "v"(ga) : "memory");
}

// ---------------------------------------------------------------------------
// One-shot fp32 -> bf16 convert (activations)
// ---------------------------------------------------------------------------
__global__ __launch_bounds__(256) void cvt_bf16(const float* __restrict__ x,
                                                unsigned short* __restrict__ y,
                                                int n) {
    const int i = (blockIdx.x * 256 + threadIdx.x) * 4;
    if (i < n) {
        const float4 v = *reinterpret_cast<const float4*>(x + i);
        y[i + 0] = f2bf(v.x); y[i + 1] = f2bf(v.y);
        y[i + 2] = f2bf(v.z); y[i + 3] = f2bf(v.w);
    }
}

// W[K][N] fp32 -> Wt[N][K] bf16 (tiled transpose through LDS)
__global__ __launch_bounds__(256) void tcvt_bf16(const float* __restrict__ W,
                                                 unsigned short* __restrict__ Wt,
                                                 int N, int K) {
    __shared__ float t[32][33];
    const int bx = blockIdx.x * 32;   // n
    const int by = blockIdx.y * 32;   // k
    const int tx = threadIdx.x, ty = threadIdx.y;
#pragma unroll
    for (int j = 0; j < 4; ++j)
        t[ty + j * 8][tx] = W[(size_t)(by + ty + j * 8) * N + bx + tx];
    __syncthreads();
#pragma unroll
    for (int j = 0; j < 4; ++j)
        Wt[(size_t)(bx + ty + j * 8) * K + by + tx] = f2bf(t[tx][ty + j * 8]);
}

// ---------------------------------------------------------------------------
// GEMM: C[M,N] = A[M,K](bf16) * Wt[N,K](bf16, pre-transposed) + bias.
// Block tile 128x128, K-step 32, 256 threads = 8 waves (4x2 wave grid),
// each wave computes 32x64. LDS double-buffered and staged with
// global_load_async_to_lds_b128 (ASYNCcnt).
// OM: 0 = fp32 [M][N]; 1 = bf16 [M][N]; 2 = bf16 per-head transposed
//     ([b*16+h][d][S], for attention V).
// ---------------------------------------------------------------------------
template <int OM>
__global__ __launch_bounds__(256) void gemm128(const unsigned short* __restrict__ A,
                                               const unsigned short* __restrict__ Wt,
                                               const float* __restrict__ bias,
                                               void* __restrict__ Cv,
                                               int M, int N, int Kd) {
    __shared__ unsigned short sA[2][128 * 32];   // [m][k]
    __shared__ unsigned short sB[2][128 * 32];   // [n][k]

    const int tid  = threadIdx.x;
    const int wave = tid >> 5;
    const int lane = tid & 31;
    const int wm   = (wave & 3) * 32;    // wave's m-strip within the block
    const int wn   = (wave >> 2) * 64;   // wave's n-strip within the block
    const int m0   = blockIdx.y * 128;
    const int n0   = blockIdx.x * 128;

    // 16B-chunk staging map: 512 chunks per 128x32 tile, 2 per thread.
    const int row0 = tid >> 2;
    const int ko0  = (tid & 3) * 8;

    v8f acc[2][4];
#pragma unroll
    for (int i = 0; i < 2; ++i)
#pragma unroll
        for (int nt = 0; nt < 4; ++nt) acc[i][nt] = 0.0f;

#define ISSUE(buf, kb)                                                                     \
    do {                                                                                   \
        async_b128((unsigned)(uintptr_t)&sA[buf][(row0)*32 + ko0],                         \
                   A + (size_t)(m0 + row0) * Kd + (kb) + ko0);                             \
        async_b128((unsigned)(uintptr_t)&sA[buf][(row0 + 64)*32 + ko0],                    \
                   A + (size_t)(m0 + row0 + 64) * Kd + (kb) + ko0);                        \
        async_b128((unsigned)(uintptr_t)&sB[buf][(row0)*32 + ko0],                         \
                   Wt + (size_t)(n0 + row0) * Kd + (kb) + ko0);                            \
        async_b128((unsigned)(uintptr_t)&sB[buf][(row0 + 64)*32 + ko0],                    \
                   Wt + (size_t)(n0 + row0 + 64) * Kd + (kb) + ko0);                       \
    } while (0)

    ISSUE(0, 0);
    int cur = 0;
    for (int kb = 0; kb < Kd; kb += 32) {
        if (kb + 32 < Kd) {
            ISSUE(cur ^ 1, kb + 32);                         // prefetch next stage
            asm volatile("s_wait_asynccnt 0x4" ::: "memory"); // current stage done
        } else {
            asm volatile("s_wait_asynccnt 0x0" ::: "memory");
        }
        __syncthreads();   // whole tile staged by all waves

        const v16bf af0 = load_a_frag(&sA[cur][(wm +  0) * 32], 32, 0, lane);
        const v16bf af1 = load_a_frag(&sA[cur][(wm + 16) * 32], 32, 0, lane);
#pragma unroll
        for (int nt = 0; nt < 4; ++nt) {
            const v16bf bf = load_b_frag(&sB[cur][(wn + nt * 16) * 32], 32, 0, lane);
            acc[0][nt] = wmma_bf16(af0, bf, acc[0][nt]);
            acc[1][nt] = wmma_bf16(af1, bf, acc[1][nt]);
        }
        __syncthreads();   // everyone done reading before buffer is re-filled
        cur ^= 1;
    }
#undef ISSUE

    // C layout: VGPR r -> row = r + (lane>=16 ? 8 : 0), col = lane&15
    const int half8 = (lane & 16) ? 8 : 0;
#pragma unroll
    for (int i = 0; i < 2; ++i) {
#pragma unroll
        for (int nt = 0; nt < 4; ++nt) {
            const int cc = n0 + wn + nt * 16 + (lane & 15);
            const float bv = bias[cc];
#pragma unroll
            for (int r = 0; r < 8; ++r) {
                const int rr = m0 + wm + i * 16 + r + half8;
                const float val = acc[i][nt][r] + bv;
                if (OM == 0) {
                    ((float*)Cv)[(size_t)rr * N + cc] = val;
                } else if (OM == 1) {
                    ((unsigned short*)Cv)[(size_t)rr * N + cc] = f2bf(val);
                } else {
                    // [b*16+h][d][S] (S=1024): B-fragment-ready layout for P@V
                    const int b = rr >> 10, s = rr & 1023;
                    const int hh = cc >> 6, dd = cc & 63;
                    ((unsigned short*)Cv)[(((size_t)b * 16 + hh) * 64 + dd) * 1024 + s] =
                        f2bf(val);
                }
            }
        }
    }
}

// ---------------------------------------------------------------------------
// Flash attention: one workgroup per (b, h, 64-query-row block), 4 waves,
// each wave owns 16 query rows and runs fully independently (no barriers).
// K ([S][d]) and Vt ([d][S]) are both B-fragment-ready in global memory.
// ---------------------------------------------------------------------------
__global__ __launch_bounds__(128) void attn64(const unsigned short* __restrict__ Q,
                                              const unsigned short* __restrict__ Kk,
                                              const unsigned short* __restrict__ Vt,
                                              const float* __restrict__ mask,
                                              unsigned short* __restrict__ O) {
    constexpr int S = 1024, H = 1024, D = 64, NHEADS = 16;
    __shared__ unsigned short Pb[4][16 * 64];     // per-wave P tile [q][s]

    const int tid  = threadIdx.x;
    const int wave = tid >> 5;
    const int lane = tid & 31;
    const int b    = blockIdx.y / NHEADS;
    const int h    = blockIdx.y % NHEADS;
    const int q0   = blockIdx.x * 64 + wave * 16;
    const int half8 = (lane & 16) ? 8 : 0;

    const unsigned short* vb = Vt + (size_t)(b * NHEADS + h) * D * S;  // [d][s]

    // ---- Q fragments straight from global in A-fragment layout ----
    v16bf qf[2];
    {
        const int row = q0 + (lane & 15);
        const int k0  = (lane & 16) ? 8 : 0;
        const unsigned short* qp = Q + ((size_t)b * S + row) * H + (size_t)h * D;
        qf[0] = cat16(*(const bf8v*)(qp + k0),      *(const bf8v*)(qp + k0 + 16));
        qf[1] = cat16(*(const bf8v*)(qp + 32 + k0), *(const bf8v*)(qp + 32 + k0 + 16));
    }

    v8f o[4];
    float mrun[8], lrun[8];
#pragma unroll
    for (int t = 0; t < 4; ++t) o[t] = 0.0f;
#pragma unroll
    for (int r = 0; r < 8; ++r) { mrun[r] = -3.0e38f; lrun[r] = 0.0f; }

    const float scale = 0.125f;                   // 1/sqrt(64)

    for (int s0 = 0; s0 < S; s0 += 64) {
        if (s0 + 64 < S) {  // prefetch next key/value block
            __builtin_prefetch(Kk + ((size_t)b * S + s0 + 64 + (lane & 15)) * H + (size_t)h * D, 0, 1);
            __builtin_prefetch(vb + (size_t)(lane & 15) * S + s0 + 64, 0, 1);
        }

        // ---- scores: S_tile = Q (16x64) x K^T (64x64) -> 4 f32 accums ----
        v8f sc[4];
#pragma unroll
        for (int t = 0; t < 4; ++t) {
            sc[t] = 0.0f;
            const int n = s0 + t * 16 + (lane & 15);
            const unsigned short* kp =
                Kk + ((size_t)b * S + n) * H + (size_t)h * D + ((lane & 16) ? 8 : 0);
            const v16bf kf0 = cat16(*(const bf8v*)(kp),      *(const bf8v*)(kp + 16));
            sc[t] = wmma_bf16(qf[0], kf0, sc[t]);
            const v16bf kf1 = cat16(*(const bf8v*)(kp + 32), *(const bf8v*)(kp + 48));
            sc[t] = wmma_bf16(qf[1], kf1, sc[t]);
        }

        // ---- additive mask bias per key column ----
        float bias_t[4];
#pragma unroll
        for (int t = 0; t < 4; ++t)
            bias_t[t] = (1.0f - mask[(size_t)b * S + s0 + t * 16 + (lane & 15)]) * -10000.0f;

        // ---- online softmax (row stats per (VGPR r, lane-half)) ----
        float p[4][8];
#pragma unroll
        for (int r = 0; r < 8; ++r) {
            float mx = sc[0][r] * scale + bias_t[0];
            p[0][r] = mx;
#pragma unroll
            for (int t = 1; t < 4; ++t) {
                p[t][r] = sc[t][r] * scale + bias_t[t];
                mx = fmaxf(mx, p[t][r]);
            }
#pragma unroll
            for (int off = 8; off >= 1; off >>= 1)
                mx = fmaxf(mx, __shfl_xor(mx, off, 16));
            const float mnew  = fmaxf(mrun[r], mx);
            const float alpha = __expf(mrun[r] - mnew);
            mrun[r] = mnew;
            float rs = 0.0f;
#pragma unroll
            for (int t = 0; t < 4; ++t) {
                p[t][r] = __expf(p[t][r] - mnew);
                rs += p[t][r];
            }
#pragma unroll
            for (int off = 8; off >= 1; off >>= 1)
                rs += __shfl_xor(rs, off, 16);
            lrun[r] = lrun[r] * alpha + rs;
#pragma unroll
            for (int t = 0; t < 4; ++t) o[t][r] *= alpha;
        }

        // ---- wave-private LDS round-trip: C-layout store -> A-layout load ----
        unsigned short* Pw = &Pb[wave][0];
#pragma unroll
        for (int t = 0; t < 4; ++t)
#pragma unroll
            for (int r = 0; r < 8; ++r)
                Pw[(r + half8) * 64 + t * 16 + (lane & 15)] = f2bf(p[t][r]);
        asm volatile("s_wait_dscnt 0x0" ::: "memory");   // lanes are lockstepped

        // ---- O += P (16x64) x V (64x64); V B-frags straight from global Vt ----
#pragma unroll
        for (int ks = 0; ks < 2; ++ks) {
            const v16bf pf = load_a_frag(Pw, 64, ks * 32, lane);
#pragma unroll
            for (int nt = 0; nt < 4; ++nt) {
                const v16bf vf =
                    load_b_frag(vb + (size_t)(nt * 16) * S, S, s0 + ks * 32, lane);
                o[nt] = wmma_bf16(pf, vf, o[nt]);
            }
        }
    }

    // ---- normalize and store attention output (bf16, [B,S,H] layout) ----
#pragma unroll
    for (int nt = 0; nt < 4; ++nt) {
        const int col = h * D + nt * 16 + (lane & 15);
#pragma unroll
        for (int r = 0; r < 8; ++r) {
            const int row = q0 + r + half8;
            const float val = o[nt][r] / lrun[r];
            O[((size_t)b * S + row) * H + col] = f2bf(val);
        }
    }
}

// ---------------------------------------------------------------------------
// Host-side launch
// ---------------------------------------------------------------------------
extern "C" void kernel_launch(void* const* d_in, const int* in_sizes, int n_in,
                              void* d_out, int out_size, void* d_ws, size_t ws_size,
                              hipStream_t stream) {
    (void)in_sizes; (void)n_in; (void)out_size; (void)ws_size;
    constexpr int B = 8, S = 1024, H = 1024;
    constexpr int M = B * S;
    const size_t MH = (size_t)M * H;   // 8Mi elements
    const size_t HH = (size_t)H * H;   // 1Mi elements

    const float* key   = (const float*)d_in[0];
    const float* value = (const float*)d_in[1];
    const float* query = (const float*)d_in[2];
    const float* mask  = (const float*)d_in[3];
    const float* Wq = (const float*)d_in[4];  const float* bq = (const float*)d_in[5];
    const float* Wk = (const float*)d_in[6];  const float* bk = (const float*)d_in[7];
    const float* Wv = (const float*)d_in[8];  const float* bv = (const float*)d_in[9];
    const float* Wo = (const float*)d_in[10]; const float* bo = (const float*)d_in[11];

    // bf16 workspace carve-out
    unsigned short* aQ  = (unsigned short*)d_ws;     // query bf16
    unsigned short* aK  = aQ  + MH;                  // key   bf16
    unsigned short* aV  = aK  + MH;                  // value bf16
    unsigned short* tWq = aV  + MH;                  // Wq^T bf16
    unsigned short* tWk = tWq + HH;
    unsigned short* tWv = tWk + HH;
    unsigned short* tWo = tWv + HH;
    unsigned short* pQ  = tWo + HH;                  // Q proj
    unsigned short* pK  = pQ  + MH;                  // K proj
    unsigned short* pVt = pK  + MH;                  // V proj, per-head [d][S]
    unsigned short* pA  = pVt + MH;                  // attention out

    // 1) one-shot conversions
    cvt_bf16<<<(int)(MH / 4 / 256), 256, 0, stream>>>(query, aQ, (int)MH);
    cvt_bf16<<<(int)(MH / 4 / 256), 256, 0, stream>>>(key,   aK, (int)MH);
    cvt_bf16<<<(int)(MH / 4 / 256), 256, 0, stream>>>(value, aV, (int)MH);
    tcvt_bf16<<<dim3(32, 32), dim3(32, 8), 0, stream>>>(Wq, tWq, H, H);
    tcvt_bf16<<<dim3(32, 32), dim3(32, 8), 0, stream>>>(Wk, tWk, H, H);
    tcvt_bf16<<<dim3(32, 32), dim3(32, 8), 0, stream>>>(Wv, tWv, H, H);
    tcvt_bf16<<<dim3(32, 32), dim3(32, 8), 0, stream>>>(Wo, tWo, H, H);

    // 2) QKV projections (async-LDS double-buffered WMMA GEMM)
    const dim3 gg(H / 128, M / 128);
    const dim3 bb(256);
    gemm128<1><<<gg, bb, 0, stream>>>(aQ, tWq, bq, pQ,  M, H, H);
    gemm128<1><<<gg, bb, 0, stream>>>(aK, tWk, bk, pK,  M, H, H);
    gemm128<2><<<gg, bb, 0, stream>>>(aV, tWv, bv, pVt, M, H, H);   // per-head [d][S]

    // 3) flash attention
    attn64<<<dim3(S / 64, B * 16), dim3(128), 0, stream>>>(pQ, pK, pVt, mask, pA);

    // 4) output projection (fp32 out + bias)
    gemm128<0><<<gg, bb, 0, stream>>>(pA, tWo, bo, (float*)d_out, M, H, H);
}
